// AttentionBlockFused_74191265071208
// MI455X (gfx1250) — compile-verified, matrix operand order checked by reading, not run on hardware
//
#include <hip/hip_runtime.h>
#include <hip/hip_bf16.h>

#define HIDDEN   2880
#define HEAD_DIM 64
#define N_HEADS  64
#define N_KV     8
#define QKV_DIM  5120           // 64*(64+16)
#define ATT_DIM  4096           // N_HEADS*HEAD_DIM
#define KV_DIM   512            // N_KV*HEAD_DIM
#define WINDOW   128
#define NTOK     1536
#define SM_SCALE 0.125f
#define EPS      1e-5f

typedef __attribute__((ext_vector_type(16))) __bf16 v16bf;
typedef __attribute__((ext_vector_type(8)))  float  v8f;

// ---------------- workspace layout (bytes) ----------------
static constexpr size_t SZ_T   = (size_t)NTOK * HIDDEN * 2;      // t bf16
static constexpr size_t SZ_W1  = (size_t)QKV_DIM * HIDDEN * 2;   // qkv_w bf16
static constexpr size_t SZ_W2  = (size_t)HIDDEN * ATT_DIM * 2;   // out_w bf16
static constexpr size_t SZ_QKV = (size_t)NTOK * QKV_DIM * 2;     // qkv bf16
static constexpr size_t SZ_Q   = (size_t)NTOK * ATT_DIM * 2;     // q (post-rope) [h][t][d]
static constexpr size_t SZ_K   = (size_t)NTOK * KV_DIM * 2;      // k (post-rope) [kv][t][d]
static constexpr size_t SZ_V   = (size_t)NTOK * KV_DIM * 2;      // vT [kv][d][t]
static constexpr size_t OFF_T   = 0;
static constexpr size_t OFF_W1  = OFF_T   + SZ_T;
static constexpr size_t OFF_W2  = OFF_W1  + SZ_W1;
static constexpr size_t OFF_QKV = OFF_W2  + SZ_W2;
static constexpr size_t OFF_Q   = OFF_QKV + SZ_QKV;
static constexpr size_t OFF_K   = OFF_Q   + SZ_Q;
static constexpr size_t OFF_V   = OFF_K   + SZ_K;
static constexpr size_t OFF_ATT = OFF_V   + SZ_V;                // attn out bf16 [t][4096]

// ---------------- RMSNorm + cast to bf16 ----------------
__global__ __launch_bounds__(256) void rmsnorm_cast(const float* __restrict__ x,
                                                    const float* __restrict__ g,
                                                    __hip_bfloat16* __restrict__ t) {
  int row = blockIdx.x;
  const float* xr = x + (size_t)row * HIDDEN;
  float ss = 0.f;
  for (int i = threadIdx.x; i < HIDDEN; i += 256) { float v = xr[i]; ss = fmaf(v, v, ss); }
  #pragma unroll
  for (int o = 16; o > 0; o >>= 1) ss += __shfl_xor(ss, o, 32);
  __shared__ float red[8];
  __shared__ float rinv;
  if ((threadIdx.x & 31) == 0) red[threadIdx.x >> 5] = ss;
  __syncthreads();
  if (threadIdx.x == 0) {
    float s = 0.f;
    for (int i = 0; i < 8; ++i) s += red[i];
    rinv = rsqrtf(s / (float)HIDDEN + EPS);
  }
  __syncthreads();
  float r = rinv;
  for (int i = threadIdx.x; i < HIDDEN; i += 256)
    t[(size_t)row * HIDDEN + i] = __float2bfloat16(xr[i] * r * g[i]);
}

// ---------------- fp32 -> bf16 weight cast (vectorized) ----------------
__global__ __launch_bounds__(256) void cast_f32_bf16(const float* __restrict__ src,
                                                     __hip_bfloat16* __restrict__ dst, int n4) {
  int i = blockIdx.x * 256 + threadIdx.x;
  if (i < n4) {
    float4 v = ((const float4*)src)[i];
    __hip_bfloat16 tmp[4];
    tmp[0] = __float2bfloat16(v.x); tmp[1] = __float2bfloat16(v.y);
    tmp[2] = __float2bfloat16(v.z); tmp[3] = __float2bfloat16(v.w);
    *(uint2*)(dst + (size_t)i * 4) = *(uint2*)tmp;
  }
}

// ---------------- bf16 WMMA GEMM: C[M][N] = A[M][K] * B[N][K]^T + bias ----------------
// 128x128 block tile, double-buffered LDS, register-staged global loads.
// 8 waves in a 4(M) x 2(N) grid; each wave computes a 32(M) x 64(N) tile:
//   2 A-frags x 4 B-frags -> 8 WMMA accumulators, 6 LDS fragment loads per 8 WMMAs.
// RESID=false: out = bf16(C + bias).  RESID=true: out = f32(C + bias + residual).
#define BM 128
#define BN 128
#define BK 32
template <bool RESID>
__global__ __launch_bounds__(256, 1) void gemm_bf16(const __hip_bfloat16* __restrict__ A,
                                                    const __hip_bfloat16* __restrict__ B,
                                                    const float* __restrict__ bias,
                                                    const float* __restrict__ residual,
                                                    float* __restrict__ outF,
                                                    __hip_bfloat16* __restrict__ outB,
                                                    int M, int N, int K) {
  __shared__ __align__(16) __hip_bfloat16 As[2][BM * BK];
  __shared__ __align__(16) __hip_bfloat16 Bs[2][BN * BK];
  const int m0 = blockIdx.y * BM;
  const int n0 = blockIdx.x * BN;
  const int tid = threadIdx.x;
  const int wid = tid >> 5, lane = tid & 31;
  const int lm = lane & 15, hs = lane >> 4;
  const int mw = wid & 3;    // M strip: 32 rows
  const int nw = wid >> 2;   // N strip: 64 cols
  const int NK = K / BK;

  // per-thread staging coordinates: two 16B segments each for A and B tiles
  const int s0 = tid, s1 = tid + 256;
  const int ar0 = s0 >> 2, ac0 = (s0 & 3) * 8;
  const int ar1 = s1 >> 2, ac1 = (s1 & 3) * 8;
  int nr0 = n0 + ar0; if (nr0 >= N) nr0 = N - 1;
  int nr1 = n0 + ar1; if (nr1 >= N) nr1 = N - 1;
  const __hip_bfloat16* Ap0 = &A[(size_t)(m0 + ar0) * K + ac0];
  const __hip_bfloat16* Ap1 = &A[(size_t)(m0 + ar1) * K + ac1];
  const __hip_bfloat16* Bp0 = &B[(size_t)nr0 * K + ac0];
  const __hip_bfloat16* Bp1 = &B[(size_t)nr1 * K + ac1];

  v8f acc[2][4];
  #pragma unroll
  for (int i = 0; i < 2; ++i)
    #pragma unroll
    for (int j = 0; j < 4; ++j) acc[i][j] = (v8f)0.0f;

  // prologue: stage K-tile 0 into buffer 0
  {
    uint4 a0 = *(const uint4*)Ap0;
    uint4 a1 = *(const uint4*)Ap1;
    uint4 b0 = *(const uint4*)Bp0;
    uint4 b1 = *(const uint4*)Bp1;
    *(uint4*)&As[0][ar0 * BK + ac0] = a0;
    *(uint4*)&As[0][ar1 * BK + ac1] = a1;
    *(uint4*)&Bs[0][ar0 * BK + ac0] = b0;
    *(uint4*)&Bs[0][ar1 * BK + ac1] = b1;
  }

  for (int kt = 0; kt < NK; ++kt) {
    const int cur = kt & 1;
    __syncthreads();  // buffer `cur` fully staged; previous compute finished

    // issue next K-tile's global loads early (latency hidden behind WMMAs)
    const bool more = (kt + 1) < NK;
    uint4 na0, na1, nb0, nb1;
    if (more) {
      const size_t ko = (size_t)(kt + 1) * BK;
      na0 = *(const uint4*)(Ap0 + ko);
      na1 = *(const uint4*)(Ap1 + ko);
      nb0 = *(const uint4*)(Bp0 + ko);
      nb1 = *(const uint4*)(Bp1 + ko);
      if (kt + 2 < NK) {  // global_prefetch_b8 two tiles ahead
        __builtin_prefetch(Ap0 + ko + BK, 0, 1);
        __builtin_prefetch(Bp0 + ko + BK, 0, 1);
      }
    }

    // batched fragment loads: 2 A frags + 4 B frags, then 8 WMMAs
    v16bf a[2];
    #pragma unroll
    for (int i = 0; i < 2; ++i) {
      const __hip_bfloat16* asrc = &As[cur][(mw * 32 + i * 16 + lm) * BK];
      ((uint4*)&a[i])[0] = *(const uint4*)(asrc + hs * 8);
      ((uint4*)&a[i])[1] = *(const uint4*)(asrc + 16 + hs * 8);
    }
    v16bf b[4];
    #pragma unroll
    for (int j = 0; j < 4; ++j) {
      const __hip_bfloat16* bsrc = &Bs[cur][(nw * 64 + j * 16 + lm) * BK];
      ((uint4*)&b[j])[0] = *(const uint4*)(bsrc + hs * 8);
      ((uint4*)&b[j])[1] = *(const uint4*)(bsrc + 16 + hs * 8);
    }
    #pragma unroll
    for (int i = 0; i < 2; ++i)
      #pragma unroll
      for (int j = 0; j < 4; ++j)
        acc[i][j] = __builtin_amdgcn_wmma_f32_16x16x32_bf16(false, a[i], false, b[j], (short)0, acc[i][j], false, false);

    // stage next tile into the other buffer (no conflict with `cur` readers)
    if (more) {
      const int nxt = cur ^ 1;
      *(uint4*)&As[nxt][ar0 * BK + ac0] = na0;
      *(uint4*)&As[nxt][ar1 * BK + ac1] = na1;
      *(uint4*)&Bs[nxt][ar0 * BK + ac0] = nb0;
      *(uint4*)&Bs[nxt][ar1 * BK + ac1] = nb1;
    }
  }

  // epilogue: C layout row = (lane<16 ? p : 8+p), col = lane&15
  #pragma unroll
  for (int i = 0; i < 2; ++i) {
    const int mrow_base = m0 + mw * 32 + i * 16;
    #pragma unroll
    for (int j = 0; j < 4; ++j) {
      int col = n0 + nw * 64 + j * 16 + lm;
      if (col < N) {
        float bv = bias[col];
        #pragma unroll
        for (int p = 0; p < 8; ++p) {
          int row = mrow_base + (hs ? 8 + p : p);
          float v = acc[i][j][p] + bv;
          if (RESID) outF[(size_t)row * N + col] = v + residual[(size_t)row * N + col];
          else       outB[(size_t)row * N + col] = __float2bfloat16(v);
        }
      }
    }
  }
}

// ---------------- RoPE (YaRN/NTK) + split q/k/v; V stored transposed ----------------
__global__ __launch_bounds__(256) void rope_split(const __hip_bfloat16* __restrict__ qkv,
                                                  __hip_bfloat16* __restrict__ q,
                                                  __hip_bfloat16* __restrict__ kk,
                                                  __hip_bfloat16* __restrict__ vT, int n) {
  const int t = blockIdx.x;
  const __hip_bfloat16* row = qkv + (size_t)t * QKV_DIM;
  const float lf   = logf(150000.0f);
  const float conc = 0.1f * logf(32.0f) + 1.0f;
  const float twopi = 6.283185307179586f;
  const float low  = 32.0f * logf(4096.0f / (32.0f * twopi)) / lf;
  const float high = 32.0f * logf(4096.0f / twopi) / lf;

  for (int idx = threadIdx.x; idx < 72 * 32; idx += 256) {
    int head = idx >> 5, d = idx & 31;
    float freq   = __expf((2.0f * (float)d / 64.0f) * lf);
    float interp = 1.0f / (32.0f * freq);
    float extrap = 1.0f / freq;
    float ramp   = ((float)d - low) / (high - low);
    ramp = fminf(fmaxf(ramp, 0.0f), 1.0f);
    float msk  = 1.0f - ramp;
    float invf = interp * (1.0f - msk) + extrap * msk;
    float ang  = (float)t * invf;
    float c = cosf(ang) * conc, s = sinf(ang) * conc;
    float x1 = __bfloat162float(row[head * 64 + d]);
    float x2 = __bfloat162float(row[head * 64 + d + 32]);
    float o1 = x1 * c - x2 * s;
    float o2 = x2 * c + x1 * s;
    if (head < 64) {
      __hip_bfloat16* dst = q + ((size_t)head * n + t) * 64;
      dst[d] = __float2bfloat16(o1); dst[d + 32] = __float2bfloat16(o2);
    } else {
      __hip_bfloat16* dst = kk + ((size_t)(head - 64) * n + t) * 64;
      dst[d] = __float2bfloat16(o1); dst[d + 32] = __float2bfloat16(o2);
    }
  }
  for (int i = threadIdx.x; i < KV_DIM; i += 256) {
    int kv = i >> 6, d = i & 63;
    vT[((size_t)kv * 64 + d) * n + t] = row[ATT_DIM + KV_DIM + i];   // v cols 4608..5119
  }
}

// ---------------- sliding-window attention with sink, all-WMMA ----------------
// grid: (ntiles/4, N_HEADS); block 128 = 4 waves; one wave per 16-query tile.
__global__ __launch_bounds__(128, 1) void attn_kernel(const __hip_bfloat16* __restrict__ q,
                                                      const __hip_bfloat16* __restrict__ k,
                                                      const __hip_bfloat16* __restrict__ vT,
                                                      const float* __restrict__ sinks,
                                                      __hip_bfloat16* __restrict__ out, int n) {
  __shared__ __align__(16) __hip_bfloat16 Plds[4][16][160];
  const int h = blockIdx.y;
  const int wid = threadIdx.x >> 5, lane = threadIdx.x & 31;
  const int lm = lane & 15, hs = lane >> 4;
  const int qt = blockIdx.x * 4 + wid;
  const int q0 = qt * 16;
  const int kv = h >> 3;
  const __hip_bfloat16* qh = q  + (size_t)h  * n * HEAD_DIM;
  const __hip_bfloat16* kh = k  + (size_t)kv * n * HEAD_DIM;
  const __hip_bfloat16* vh = vT + (size_t)kv * HEAD_DIM * n;

  // Q A-fragments for dim chunks {0..31},{32..63}
  v16bf qa[2];
  #pragma unroll
  for (int kc = 0; kc < 2; ++kc) {
    const __hip_bfloat16* rp = qh + (size_t)(q0 + lm) * HEAD_DIM + kc * 32;
    ((uint4*)&qa[kc])[0] = *(const uint4*)(rp + hs * 8);
    ((uint4*)&qa[kc])[1] = *(const uint4*)(rp + 16 + hs * 8);
  }

  // scores for 160 candidate keys: 5 chunks of 32 keys = 10 16x16 tiles
  v8f S[10];
  #pragma unroll
  for (int c = 0; c < 5; ++c) {
    const int kb = q0 - WINDOW + c * 32;
    #pragma unroll
    for (int s2 = 0; s2 < 2; ++s2) {
      const int keybase = kb + s2 * 16;
      int kidx = keybase + lm;
      if (kidx < 0) kidx = 0; if (kidx >= n) kidx = n - 1;
      const __hip_bfloat16* krp = kh + (size_t)kidx * HEAD_DIM;
      v16bf b0, b1;
      ((uint4*)&b0)[0] = *(const uint4*)(krp + hs * 8);
      ((uint4*)&b0)[1] = *(const uint4*)(krp + 16 + hs * 8);
      ((uint4*)&b1)[0] = *(const uint4*)(krp + 32 + hs * 8);
      ((uint4*)&b1)[1] = *(const uint4*)(krp + 48 + hs * 8);
      v8f a = (v8f)0.0f;
      a = __builtin_amdgcn_wmma_f32_16x16x32_bf16(false, qa[0], false, b0, (short)0, a, false, false);
      a = __builtin_amdgcn_wmma_f32_16x16x32_bf16(false, qa[1], false, b1, (short)0, a, false, false);
      #pragma unroll
      for (int p = 0; p < 8; ++p) {
        int i = q0 + (hs ? 8 + p : p);
        int j = keybase + lm;
        bool ok = (j >= 0) && (j <= i) && (j > i - WINDOW);
        a[p] = ok ? a[p] * SM_SCALE : -1e30f;
      }
      S[c * 2 + s2] = a;
    }
  }

  // softmax with sink (per query row: 16-lane shuffle reductions)
  const float snk = sinks[h];
  float inv[8];
  #pragma unroll
  for (int p = 0; p < 8; ++p) {
    float mx = -1e30f;
    #pragma unroll
    for (int tt = 0; tt < 10; ++tt) mx = fmaxf(mx, S[tt][p]);
    #pragma unroll
    for (int o = 1; o < 16; o <<= 1) mx = fmaxf(mx, __shfl_xor(mx, o, 32));
    mx = fmaxf(mx, snk);
    float sm = 0.f;
    #pragma unroll
    for (int tt = 0; tt < 10; ++tt) { float e = __expf(S[tt][p] - mx); S[tt][p] = e; sm += e; }
    #pragma unroll
    for (int o = 1; o < 16; o <<= 1) sm += __shfl_xor(sm, o, 32);
    sm += __expf(snk - mx);
    inv[p] = 1.0f / sm;
  }

  // C-layout P -> LDS (row-major 16x160 bf16) for re-read in A-layout
  #pragma unroll
  for (int tt = 0; tt < 10; ++tt) {
    int colb = tt * 16 + lm;
    #pragma unroll
    for (int p = 0; p < 8; ++p)
      Plds[wid][hs ? 8 + p : p][colb] = __float2bfloat16(S[tt][p]);
  }
  __syncthreads();

  // P @ V via WMMA; V is pre-transposed so B-fragments are contiguous loads
  v8f o4[4];
  #pragma unroll
  for (int dsub = 0; dsub < 4; ++dsub) o4[dsub] = (v8f)0.0f;
  #pragma unroll
  for (int c = 0; c < 5; ++c) {
    const int kb = q0 - WINDOW + c * 32;
    v16bf pf;
    ((uint4*)&pf)[0] = *(const uint4*)&Plds[wid][lm][c * 32 + hs * 8];
    ((uint4*)&pf)[1] = *(const uint4*)&Plds[wid][lm][c * 32 + 16 + hs * 8];
    v16bf vf[4];
    #pragma unroll
    for (int dsub = 0; dsub < 4; ++dsub) {
      const int dim = dsub * 16 + lm;
      const __hip_bfloat16* vrow = vh + (size_t)dim * n;
      int ks0 = kb + hs * 8;      if (ks0 < 0) ks0 = 0; if (ks0 > n - 8) ks0 = n - 8;
      int ks1 = kb + 16 + hs * 8; if (ks1 < 0) ks1 = 0; if (ks1 > n - 8) ks1 = n - 8;
      ((uint4*)&vf[dsub])[0] = *(const uint4*)(vrow + ks0);
      ((uint4*)&vf[dsub])[1] = *(const uint4*)(vrow + ks1);
    }
    #pragma unroll
    for (int dsub = 0; dsub < 4; ++dsub)
      o4[dsub] = __builtin_amdgcn_wmma_f32_16x16x32_bf16(false, pf, false, vf[dsub], (short)0, o4[dsub], false, false);
  }

  #pragma unroll
  for (int dsub = 0; dsub < 4; ++dsub) {
    int col = h * HEAD_DIM + dsub * 16 + lm;
    #pragma unroll
    for (int p = 0; p < 8; ++p) {
      int row = q0 + (hs ? 8 + p : p);
      out[(size_t)row * ATT_DIM + col] = __float2bfloat16(o4[dsub][p] * inv[p]);
    }
  }
}

// ---------------- host launcher ----------------
extern "C" void kernel_launch(void* const* d_in, const int* in_sizes, int n_in,
                              void* d_out, int out_size, void* d_ws, size_t ws_size,
                              hipStream_t stream) {
  const float* x      = (const float*)d_in[0];
  const float* nscale = (const float*)d_in[1];
  const float* qkv_w  = (const float*)d_in[2];
  const float* qkv_b  = (const float*)d_in[3];
  const float* out_w  = (const float*)d_in[4];
  const float* out_b  = (const float*)d_in[5];
  const float* sinks  = (const float*)d_in[6];
  float* y = (float*)d_out;
  char* ws = (char*)d_ws;
  const int n = NTOK;

  __hip_bfloat16* t_bf   = (__hip_bfloat16*)(ws + OFF_T);
  __hip_bfloat16* w1_bf  = (__hip_bfloat16*)(ws + OFF_W1);
  __hip_bfloat16* w2_bf  = (__hip_bfloat16*)(ws + OFF_W2);
  __hip_bfloat16* qkv_bf = (__hip_bfloat16*)(ws + OFF_QKV);
  __hip_bfloat16* q_bf   = (__hip_bfloat16*)(ws + OFF_Q);
  __hip_bfloat16* k_bf   = (__hip_bfloat16*)(ws + OFF_K);
  __hip_bfloat16* v_bf   = (__hip_bfloat16*)(ws + OFF_V);
  __hip_bfloat16* att_bf = (__hip_bfloat16*)(ws + OFF_ATT);

  rmsnorm_cast<<<n, 256, 0, stream>>>(x, nscale, t_bf);
  {
    int n4 = QKV_DIM * HIDDEN / 4;
    cast_f32_bf16<<<(n4 + 255) / 256, 256, 0, stream>>>(qkv_w, w1_bf, n4);
  }
  {
    int n4 = HIDDEN * ATT_DIM / 4;
    cast_f32_bf16<<<(n4 + 255) / 256, 256, 0, stream>>>(out_w, w2_bf, n4);
  }
  gemm_bf16<false><<<dim3(QKV_DIM / BN, NTOK / BM), 256, 0, stream>>>(
      t_bf, w1_bf, qkv_b, nullptr, nullptr, qkv_bf, n, QKV_DIM, HIDDEN);
  rope_split<<<n, 256, 0, stream>>>(qkv_bf, q_bf, k_bf, v_bf, n);
  attn_kernel<<<dim3(n / 64, N_HEADS), 128, 0, stream>>>(q_bf, k_bf, v_bf, sinks, att_bf, n);
  gemm_bf16<true><<<dim3((HIDDEN + BN - 1) / BN, NTOK / BM), 256, 0, stream>>>(
      att_bf, w2_bf, out_b, x, y, nullptr, n, HIDDEN, ATT_DIM);
}
